// MultiheadAttention_61684320305473
// MI455X (gfx1250) — matrix operand order repaired and checked by reading, not compiled, not run on hardware
//
#include <hip/hip_runtime.h>

#define USE_ASYNC 1

// ---------------- problem constants ----------------
#define SEQ    4096
#define DM     512
#define NH     8
#define DH     64

typedef __attribute__((ext_vector_type(16))) __bf16 v16bf;
typedef __attribute__((ext_vector_type(8)))  __bf16 v8bf;
typedef __attribute__((ext_vector_type(8)))  float  v8f;

// ---------------- fragment helpers (CDNA5 WMMA layouts, ISA 7.12.2) ----------
// A-frag (16x32 bf16): lane holds row M=lane%16; kb = 8*(lane/16);
//   elements 0..7  = K kb..kb+7, elements 8..15 = K 16+kb..16+kb+7
__device__ inline v16bf frag_2chunks(const __bf16* p0, const __bf16* p1) {
  v8bf a = *(const v8bf*)p0;
  v8bf b = *(const v8bf*)p1;
  v16bf r;
#pragma unroll
  for (int i = 0; i < 8; ++i) { r[i] = a[i]; r[i + 8] = b[i]; }
  return r;
}

__device__ inline v8f zero8() {
  v8f z;
#pragma unroll
  for (int i = 0; i < 8; ++i) z[i] = 0.0f;
  return z;
}

// ---------------- DPP 16-lane butterfly reductions (no DS, no waits) --------
template <int CTRL>
__device__ inline float dpp_mov(float v) {
  int i = __builtin_bit_cast(int, v);
  int m = __builtin_amdgcn_update_dpp(0, i, CTRL, 0xf, 0xf, true);
  return __builtin_bit_cast(float, m);
}
__device__ inline float red16_max(float v) {
  v = fmaxf(v, dpp_mov<0xB1>(v));   // quad_perm(1,0,3,2)
  v = fmaxf(v, dpp_mov<0x4E>(v));   // quad_perm(2,3,0,1)
  v = fmaxf(v, dpp_mov<0x141>(v));  // row_half_mirror
  v = fmaxf(v, dpp_mov<0x140>(v));  // row_mirror
  return v;
}
__device__ inline float red16_sum(float v) {
  v += dpp_mov<0xB1>(v);
  v += dpp_mov<0x4E>(v);
  v += dpp_mov<0x141>(v);
  v += dpp_mov<0x140>(v);
  return v;
}

// ---------------- conversions ----------------
__global__ void cvt_f32_bf16(const float* __restrict__ src,
                             __bf16* __restrict__ dst, int n) {
  int i = blockIdx.x * blockDim.x + threadIdx.x;
  if (i < n) dst[i] = (__bf16)src[i];
}

// W [NH][DM][DH] (f32) -> WT [NH][DH][DM] (bf16)
__global__ void cvt_w_T(const float* __restrict__ src, __bf16* __restrict__ dst) {
  int i = blockIdx.x * blockDim.x + threadIdx.x;  // NH*DM*DH
  int e = i % DH;
  int d = (i / DH) % DM;
  int h = i / (DH * DM);
  dst[((size_t)h * DH + e) * DM + d] = (__bf16)src[i];
}

// Wo [DM][DM] (f32) -> WoT transposed (bf16)
__global__ void cvt_wo_T(const float* __restrict__ src, __bf16* __restrict__ dst) {
  int i = blockIdx.x * blockDim.x + threadIdx.x;  // DM*DM
  int n = i % DM;
  int d = i / DM;
  dst[(size_t)n * DM + d] = (__bf16)src[i];
}

// ---------------- Q/K projection (swapped orientation) ----------------
// Computes D[e][s] = WT[h] (A, 16x32 over d) x x^T (B, rows of xb are
// contiguous K) so ALL loads are contiguous and the C-layout gives 8
// consecutive e per lane -> one v8bf store per tile.
// grid (SEQ/128, NH), block 256; wave w owns s-strip of 16, full e=64.
__global__ __launch_bounds__(256) void qk_proj(
    const __bf16* __restrict__ xb,    // [SEQ][DM]
    const __bf16* __restrict__ WTb,   // [NH][DH][DM]
    const float* __restrict__ bias,   // [NH][DH]
    float scale,
    __bf16* __restrict__ outb) {      // [NH][SEQ][DH]
  const int h     = blockIdx.y;
  const int sbase = blockIdx.x * 128;
  const int wv    = threadIdx.x >> 5;
  const int lane  = threadIdx.x & 31;
  const int half  = lane >> 4;
  const int l16   = lane & 15;
  const int kb    = half * 8;
  const int scol  = sbase + wv * 16 + l16;  // this lane's s column

  v8f acc[4];
#pragma unroll
  for (int et = 0; et < 4; ++et) acc[et] = zero8();

  for (int ks = 0; ks < DM; ks += 32) {
    // B-frag: lane = col s, d chunks kb..kb+7 and 16+kb..16+kb+7
    v16bf b = frag_2chunks(xb + (size_t)scol * DM + ks + kb, xb + (size_t)scol * DM + ks + kb + 16);
#pragma unroll
    for (int et = 0; et < 4; ++et) {
      // A-frag: lane = row e, d-chunks contiguous in WT
      const __bf16* ap =
          WTb + ((size_t)h * DH + et * 16 + l16) * DM + ks + kb;
      v16bf a = frag_2chunks(ap, ap + 16);
      acc[et] = __builtin_amdgcn_wmma_f32_16x16x32_bf16(
          false, a, false, b, (short)0, acc[et], false, false);
    }
  }
#pragma unroll
  for (int et = 0; et < 4; ++et) {
    int e0 = et * 16 + 8 * half;  // 8 consecutive e values in this lane
    v8f bch = *(const v8f*)(bias + h * DH + e0);
    v8bf pk;
#pragma unroll
    for (int r = 0; r < 8; ++r) pk[r] = (__bf16)((acc[et][r] + bch[r]) * scale);
    *(v8bf*)(outb + ((size_t)h * SEQ + scol) * DH + e0) = pk;
  }
}

// ---------------- V projection (original orientation, transposed store) ------
// D[s][e]; the 8 consecutive s per lane vector-store into VT[e][s].
__global__ __launch_bounds__(256) void v_proj(
    const __bf16* __restrict__ xb,    // [SEQ][DM]
    const __bf16* __restrict__ WTb,   // [NH][DH][DM]
    const float* __restrict__ bias,   // [NH][DH]
    __bf16* __restrict__ outT) {      // [NH][DH][SEQ]
  const int h     = blockIdx.y;
  const int mbase = blockIdx.x * 128;
  const int wv    = threadIdx.x >> 5;
  const int lane  = threadIdx.x & 31;
  const int half  = lane >> 4;
  const int l16   = lane & 15;
  const int kb    = half * 8;
  const int row   = mbase + wv * 16 + l16;

  v8f acc[4];
#pragma unroll
  for (int nt = 0; nt < 4; ++nt) acc[nt] = zero8();

  for (int ks = 0; ks < DM; ks += 32) {
    const __bf16* ap = xb + (size_t)row * DM + ks + kb;
    v16bf a = frag_2chunks(ap, ap + 16);
#pragma unroll
    for (int nt = 0; nt < 4; ++nt) {
      const __bf16* bp = WTb + ((size_t)h * DH + nt * 16 + l16) * DM + ks + kb;
      v16bf b = frag_2chunks(bp, bp + 16);
      acc[nt] = __builtin_amdgcn_wmma_f32_16x16x32_bf16(
          false, a, false, b, (short)0, acc[nt], false, false);
    }
  }
#pragma unroll
  for (int nt = 0; nt < 4; ++nt) {
    int e = nt * 16 + l16;
    float bv = bias[h * DH + e];
    v8bf pk;
#pragma unroll
    for (int r = 0; r < 8; ++r) pk[r] = (__bf16)(acc[nt][r] + bv);
    *(v8bf*)(outT + ((size_t)h * DH + e) * SEQ + mbase + wv * 16 + 8 * half) = pk;
  }
}

// ---------------- Flash attention per head ----------------
__global__ __launch_bounds__(256) void attn_flash(
    const __bf16* __restrict__ Qb,  // [NH][SEQ][DH]  (pre-scaled by 1/sqrt(DH))
    const __bf16* __restrict__ Kb,  // [NH][SEQ][DH]
    const __bf16* __restrict__ VT,  // [NH][DH][SEQ]  (transposed)
    __bf16* __restrict__ Hb) {      // [SEQ][DM] concat layout
  __shared__ __attribute__((aligned(32))) __bf16 Kt[2][64 * DH];
  __shared__ __attribute__((aligned(32))) __bf16 VtT[2][DH * 64];
  __shared__ __attribute__((aligned(32))) __bf16 Pt[8][16 * 64];

  const int h     = blockIdx.y;
  const int qbase = blockIdx.x * 128;
  const int wv    = threadIdx.x >> 5;
  const int lane  = threadIdx.x & 31;
  const int half  = lane >> 4;
  const int l16   = lane & 15;
  const int kb    = half * 8;
  const int qrow  = qbase + wv * 16 + l16;

  const __bf16* Qh = Qb + (size_t)h * SEQ * DH;
  const __bf16* Kh = Kb + (size_t)h * SEQ * DH;
  const __bf16* Vh = VT + (size_t)h * DH * SEQ;

  v16bf qf[2];
#pragma unroll
  for (int es = 0; es < 2; ++es) {
    const __bf16* ap = Qh + (size_t)qrow * DH + es * 32 + kb;
    qf[es] = frag_2chunks(ap, ap + 16);
  }

  v8f o[4];
#pragma unroll
  for (int et = 0; et < 4; ++et) o[et] = zero8();
  float m8[8], l8[8];
#pragma unroll
  for (int r = 0; r < 8; ++r) { m8[r] = -1e30f; l8[r] = 0.0f; }

  const int sidx = threadIdx.x * 16;
  const int ve   = sidx >> 6;
  const int vto  = sidx & 63;

  auto issue_tile = [&](int t0, int buf) {
    const __bf16* gk = Kh + (size_t)t0 * DH + sidx;
    const __bf16* gv = Vh + (size_t)ve * SEQ + t0 + vto;
#if USE_ASYNC
    unsigned int kl = (unsigned int)(uintptr_t)(Kt[buf] + sidx);
    unsigned int vl = (unsigned int)(uintptr_t)(VtT[buf] + sidx);
    unsigned long long kg = (unsigned long long)(uintptr_t)gk;
    unsigned long long vg = (unsigned long long)(uintptr_t)gv;
    asm volatile(
        "global_load_async_to_lds_b128 %0, %2, off\n\t"
        "global_load_async_to_lds_b128 %0, %2, off offset:16\n\t"
        "global_load_async_to_lds_b128 %1, %3, off\n\t"
        "global_load_async_to_lds_b128 %1, %3, off offset:16"
        :: "v"(kl), "v"(vl), "v"(kg), "v"(vg)
        : "memory");
#else
    *(v8bf*)(Kt[buf] + sidx)      = *(const v8bf*)gk;
    *(v8bf*)(Kt[buf] + sidx + 8)  = *(const v8bf*)(gk + 8);
    *(v8bf*)(VtT[buf] + sidx)     = *(const v8bf*)gv;
    *(v8bf*)(VtT[buf] + sidx + 8) = *(const v8bf*)(gv + 8);
#endif
  };

  issue_tile(0, 0);

  for (int it = 0; it < SEQ / 64; ++it) {
    const int buf = it & 1;
    __syncthreads();  // everyone done with previous iteration
    if (it + 1 < SEQ / 64) {
      issue_tile((it + 1) * 64, buf ^ 1);
#if USE_ASYNC
      asm volatile("s_wait_asynccnt 0x4" ::: "memory");  // current tile landed
#endif
    } else {
#if USE_ASYNC
      asm volatile("s_wait_asynccnt 0x0" ::: "memory");
#endif
    }
    __syncthreads();

    const __bf16* Kc = Kt[buf];
    const __bf16* Vc = VtT[buf];

    // S = Q @ K^T : keep A operand stable across consecutive WMMAs
    v8f s[4];
#pragma unroll
    for (int nt = 0; nt < 4; ++nt) s[nt] = zero8();
#pragma unroll
    for (int es = 0; es < 2; ++es)
#pragma unroll
      for (int nt = 0; nt < 4; ++nt) {
        const __bf16* bp = Kc + (nt * 16 + l16) * DH + es * 32 + kb;
        v16bf b = frag_2chunks(bp, bp + 16);
        s[nt] = __builtin_amdgcn_wmma_f32_16x16x32_bf16(
            false, qf[es], false, b, (short)0, s[nt], false, false);
      }

    // online softmax; DPP butterfly within each 16-lane half
    float alpha[8];
#pragma unroll
    for (int r = 0; r < 8; ++r) {
      float rm = fmaxf(fmaxf(s[0][r], s[1][r]), fmaxf(s[2][r], s[3][r]));
      rm = red16_max(rm);
      float mn = fmaxf(m8[r], rm);
      alpha[r] = __expf(m8[r] - mn);
      m8[r] = mn;
    }
#pragma unroll
    for (int nt = 0; nt < 4; ++nt)
#pragma unroll
      for (int r = 0; r < 8; ++r) s[nt][r] = __expf(s[nt][r] - m8[r]);
#pragma unroll
    for (int r = 0; r < 8; ++r) {
      float rs = (s[0][r] + s[1][r]) + (s[2][r] + s[3][r]);
      rs = red16_sum(rs);
      l8[r] = l8[r] * alpha[r] + rs;
    }
#pragma unroll
    for (int et = 0; et < 4; ++et)
#pragma unroll
      for (int r = 0; r < 8; ++r) o[et][r] *= alpha[r];

    // transpose P (lane=col) -> A-layout (lane=row) via per-wave LDS tile
    __bf16* pw = Pt[wv];
#pragma unroll
    for (int nt = 0; nt < 4; ++nt)
#pragma unroll
      for (int r = 0; r < 8; ++r)
        pw[(r + 8 * half) * 64 + nt * 16 + l16] = (__bf16)s[nt][r];
    __syncthreads();

    v16bf pf[2];
#pragma unroll
    for (int ks2 = 0; ks2 < 2; ++ks2) {
      const __bf16* ap = pw + l16 * 64 + ks2 * 32 + kb;
      pf[ks2] = frag_2chunks(ap, ap + 16);
    }

    // O += P @ V : A operand stable across the inner 4 WMMAs
#pragma unroll
    for (int ks2 = 0; ks2 < 2; ++ks2)
#pragma unroll
      for (int et = 0; et < 4; ++et) {
        const __bf16* bp = Vc + (et * 16 + l16) * 64 + ks2 * 32 + kb;
        v16bf b = frag_2chunks(bp, bp + 16);
        o[et] = __builtin_amdgcn_wmma_f32_16x16x32_bf16(
            false, pf[ks2], false, b, (short)0, o[et], false, false);
      }
  }

  // finalize and store concat layout Hb[s][h*64 + e]
#pragma unroll
  for (int et = 0; et < 4; ++et) {
    int e = et * 16 + l16;
#pragma unroll
    for (int r = 0; r < 8; ++r) {
      int m = qbase + wv * 16 + r + 8 * half;
      Hb[(size_t)m * DM + h * DH + e] = (__bf16)(o[et][r] / l8[r]);
    }
  }
}

// ---------------- output projection (swapped orientation) ----------------
// D[n][s] = WoT (A) x H^T (B: rows of Hb contiguous).  Stores are one v8f
// (32B) per tile per lane; bias loads as a v8f chunk.
// grid (SEQ/128, DM/64), block 256.
__global__ __launch_bounds__(256) void out_proj(
    const __bf16* __restrict__ Hb,   // [SEQ][DM]
    const __bf16* __restrict__ WoT,  // [DM][DM] transposed
    const float* __restrict__ bo,    // [DM]
    float* __restrict__ out) {       // [SEQ][DM]
  const int sbase = blockIdx.x * 128;
  const int nbase = blockIdx.y * 64;
  const int wv    = threadIdx.x >> 5;
  const int lane  = threadIdx.x & 31;
  const int half  = lane >> 4;
  const int l16   = lane & 15;
  const int kb    = half * 8;
  const int scol  = sbase + wv * 16 + l16;

  v8f acc[4];
#pragma unroll
  for (int et = 0; et < 4; ++et) acc[et] = zero8();

  for (int ks = 0; ks < DM; ks += 32) {
    v16bf b = frag_2chunks(Hb + (size_t)scol * DM + ks + kb, Hb + (size_t)scol * DM + ks + kb + 16);
#pragma unroll
    for (int et = 0; et < 4; ++et) {
      const __bf16* ap =
          WoT + (size_t)(nbase + et * 16 + l16) * DM + ks + kb;
      v16bf a = frag_2chunks(ap, ap + 16);
      acc[et] = __builtin_amdgcn_wmma_f32_16x16x32_bf16(
          false, a, false, b, (short)0, acc[et], false, false);
    }
  }
#pragma unroll
  for (int et = 0; et < 4; ++et) {
    int n0 = nbase + et * 16 + 8 * half;  // 8 consecutive n in this lane
    v8f bch = *(const v8f*)(bo + n0);
    v8f ov;
#pragma unroll
    for (int r = 0; r < 8; ++r) ov[r] = acc[et][r] + bch[r];
    *(v8f*)(out + (size_t)scol * DM + n0) = ov;
  }
}

// ---------------- launcher ----------------
extern "C" void kernel_launch(void* const* d_in, const int* in_sizes, int n_in,
                              void* d_out, int out_size, void* d_ws,
                              size_t ws_size, hipStream_t stream) {
  const float* x  = (const float*)d_in[0];
  const float* Wq = (const float*)d_in[1];
  const float* bq = (const float*)d_in[2];
  const float* Wk = (const float*)d_in[3];
  const float* bk = (const float*)d_in[4];
  const float* Wv = (const float*)d_in[5];
  const float* bv = (const float*)d_in[6];
  const float* Wo = (const float*)d_in[7];
  const float* bo = (const float*)d_in[8];
  float* out = (float*)d_out;

  char* p = (char*)d_ws;
  __bf16* xb  = (__bf16*)p; p += (size_t)SEQ * DM * 2;
  __bf16* WqT = (__bf16*)p; p += (size_t)NH * DM * DH * 2;
  __bf16* WkT = (__bf16*)p; p += (size_t)NH * DM * DH * 2;
  __bf16* WvT = (__bf16*)p; p += (size_t)NH * DM * DH * 2;
  __bf16* WoT = (__bf16*)p; p += (size_t)DM * DM * 2;
  __bf16* Qb  = (__bf16*)p; p += (size_t)NH * SEQ * DH * 2;
  __bf16* Kb  = (__bf16*)p; p += (size_t)NH * SEQ * DH * 2;
  __bf16* VT  = (__bf16*)p; p += (size_t)NH * SEQ * DH * 2;
  __bf16* Hb  = (__bf16*)p; p += (size_t)SEQ * DM * 2;

  const int nx = SEQ * DM;
  const int nw = NH * DM * DH;
  const int no = DM * DM;
  cvt_f32_bf16<<<(nx + 255) / 256, 256, 0, stream>>>(x, xb, nx);
  cvt_w_T<<<(nw + 255) / 256, 256, 0, stream>>>(Wq, WqT);
  cvt_w_T<<<(nw + 255) / 256, 256, 0, stream>>>(Wk, WkT);
  cvt_w_T<<<(nw + 255) / 256, 256, 0, stream>>>(Wv, WvT);
  cvt_wo_T<<<(no + 255) / 256, 256, 0, stream>>>(Wo, WoT);

  dim3 gp(SEQ / 128, NH);
  const float qscale = 0.125f;  // 1/sqrt(DH), folded into Q (exact in bf16)
  qk_proj<<<gp, 256, 0, stream>>>(xb, WqT, bq, qscale, Qb);
  qk_proj<<<gp, 256, 0, stream>>>(xb, WkT, bk, 1.0f, Kb);
  v_proj<<<gp, 256, 0, stream>>>(xb, WvT, bv, VT);

  attn_flash<<<dim3(SEQ / 128, NH), 256, 0, stream>>>(Qb, Kb, VT, Hb);

  out_proj<<<dim3(SEQ / 128, DM / 64), 256, 0, stream>>>(Hb, WoT, bo, out);
}
